// MoE_2216203125013
// MI455X (gfx1250) — matrix operand order repaired and checked by reading, not compiled, not run on hardware
//
#include <hip/hip_runtime.h>
#include <hip/hip_bf16.h>

typedef __bf16 bf16;
typedef __attribute__((ext_vector_type(16))) __bf16 v16bf;
typedef __attribute__((ext_vector_type(8)))  __bf16 v8bf;
typedef __attribute__((ext_vector_type(4)))  __bf16 v4bf;
typedef __attribute__((ext_vector_type(8)))  float  v8f;

#define N_TOK 8192
#define DDIM  1024
#define IDIM  2048
#define NEXP  8
#define MT    128
#define NT    128
#define LDSK  (DDIM + 8)   // padded LDS row stride (elements) -> conflict-free A loads

// CDNA5 async global->LDS copy, 16B per lane, tracked by ASYNCcnt.
// The LDS destination operand is the low 32 bits of the generic pointer
// (= LDS byte offset); taking the address also makes the shared object
// escape so the compiler keeps the subsequent ds_loads honest.
__device__ __forceinline__ void async_g2l_b128(const void* gsrc, void* ldst)
{
    unsigned loff = (unsigned)(unsigned long long)(size_t)ldst;
    asm volatile("global_load_async_to_lds_b128 %0, %1, off"
                 :: "v"(loff), "v"(gsrc) : "memory");
}

// -------- Kernel 1: gating scores, top-4 mask * top1 weight, x -> bf16 --------
__global__ void moe_gate_topk_cvt(const float* __restrict__ x,
                                  const float* __restrict__ gw,
                                  bf16* __restrict__ xh,
                                  float* __restrict__ fm)
{
    const int lane = threadIdx.x & 31;
    const int wv   = threadIdx.x >> 5;
    const int tok  = blockIdx.x * 8 + wv;          // one wave per token
    const float* xr = x + (size_t)tok * DDIM;

    float s[NEXP];
#pragma unroll
    for (int e = 0; e < NEXP; ++e) s[e] = 0.f;

    for (int d = lane; d < DDIM; d += 32) {
        float xv = xr[d];
        xh[(size_t)tok * DDIM + d] = (bf16)xv;     // fused f32 -> bf16 convert
#pragma unroll
        for (int e = 0; e < NEXP; ++e) s[e] += xv * gw[e * DDIM + d];
    }
    // butterfly reduce: every lane ends with full dot products
#pragma unroll
    for (int off = 16; off > 0; off >>= 1) {
#pragma unroll
        for (int e = 0; e < NEXP; ++e) s[e] += __shfl_xor(s[e], off, 32);
    }
    if (lane < NEXP) {
        const int e = lane;
        float top1 = s[0];
#pragma unroll
        for (int j = 1; j < NEXP; ++j) top1 = fmaxf(top1, s[j]);
        int rank = 0;                              // stable rank (ties broken by index)
#pragma unroll
        for (int j = 0; j < NEXP; ++j)
            rank += ((s[j] > s[e]) || (s[j] == s[e] && j < e)) ? 1 : 0;
        fm[tok * NEXP + e] = (rank < 4) ? top1 : 0.f;
    }
}

// -------- Kernel 2: bulk f32 -> bf16 convert (weights) --------
__global__ void cvt_f32_bf16(const float* __restrict__ in, bf16* __restrict__ out, int n4)
{
    int i = blockIdx.x * blockDim.x + threadIdx.x;
    if (i < n4) {
        float4 v = ((const float4*)in)[i];
        v4bf o = { (bf16)v.x, (bf16)v.y, (bf16)v.z, (bf16)v.w };
        ((v4bf*)out)[i] = o;
    }
}

// -------- Kernel 3: dense 8-expert WMMA GEMM, masked accumulate, relu^2 -> bf16 H --------
__global__ __launch_bounds__(256)
void moe_expert_gemm(const bf16* __restrict__ xh,
                     const bf16* __restrict__ ewh,
                     const float* __restrict__ fm,
                     bf16* __restrict__ H)
{
    __shared__ bf16 xs[MT * LDSK];                 // 128 x (1024+8) bf16 = 264 KB of 320 KB

    const int tid  = threadIdx.x;
    const int lane = tid & 31;
    const int wv   = tid >> 5;                     // 8 waves
    const int wm   = wv >> 1;                      // 0..3 : 32-row group
    const int wn   = wv & 1;                       // 0..1 : 64-col group
    const int i0   = blockIdx.x * NT;
    const int m0   = blockIdx.y * MT;

    // ---- stage full-K X tile into LDS with CDNA5 async copies (ASYNCcnt) ----
    for (int c = tid; c < (MT * DDIM) / 8; c += 256) {
        int row  = c >> 7;                         // DDIM/8 == 128 chunks per row
        int col8 = c & 127;
        async_g2l_b128(xh + (size_t)(m0 + row) * DDIM + col8 * 8,
                       xs + row * LDSK + col8 * 8);
    }
    asm volatile("s_wait_asynccnt 0x0" ::: "memory");
    __syncthreads();

    const int rsel  = lane & 15;
    const int hi    = lane >> 4;
    const int koffA = hi ? 8 : 0;                  // A: K runs {0-7,16-23} / {8-15,24-31}
    const int koffB = hi ? 8 : 0;                  // B: K runs {0-7,16-23} / {8-15,24-31}

    v8f tot[2][4];
#pragma unroll
    for (int t = 0; t < 2; ++t)
#pragma unroll
        for (int u = 0; u < 4; ++u)
            tot[t][u] = (v8f){0,0,0,0,0,0,0,0};

    for (int e = 0; e < NEXP; ++e) {
        const bf16* We = ewh + (size_t)e * IDIM * DDIM;
        if (e + 1 < NEXP)   // pre-touch next expert's B tile rows -> global_prefetch_b8
            __builtin_prefetch(ewh + (size_t)(e + 1) * IDIM * DDIM
                                   + (size_t)(i0 + wn * 64 + rsel) * DDIM, 0, 1);

        v8f acc[2][4];
#pragma unroll
        for (int t = 0; t < 2; ++t)
#pragma unroll
            for (int u = 0; u < 4; ++u)
                acc[t][u] = (v8f){0,0,0,0,0,0,0,0};

        for (int kb = 0; kb < DDIM; kb += 32) {
            v16bf a[2];
#pragma unroll
            for (int t = 0; t < 2; ++t) {
                const bf16* p = xs + (wm * 32 + t * 16 + rsel) * LDSK + kb + koffA;
                v8bf lo = *(const v8bf*)p;
                v8bf hh = *(const v8bf*)(p + 16);
                a[t] = __builtin_shufflevector(lo, hh, 0,1,2,3,4,5,6,7,8,9,10,11,12,13,14,15);
            }
            v16bf b[4];
#pragma unroll
            for (int u = 0; u < 4; ++u) {
                const bf16* p = We + (size_t)(i0 + wn * 64 + u * 16 + rsel) * DDIM + kb + koffB;
                v8bf blo = *(const v8bf*)p;
                v8bf bhi = *(const v8bf*)(p + 16);
                b[u] = __builtin_shufflevector(blo, bhi, 0,1,2,3,4,5,6,7,8,9,10,11,12,13,14,15);
            }
#pragma unroll
            for (int t = 0; t < 2; ++t)
#pragma unroll
                for (int u = 0; u < 4; ++u)
                    acc[t][u] = __builtin_amdgcn_wmma_f32_16x16x32_bf16(
                        false, a[t], false, b[u], (short)0, acc[t][u], false, false);
        }

        // masked accumulate: tot += f[row,e] * acc   (f = top1 if expert selected)
#pragma unroll
        for (int t = 0; t < 2; ++t) {
            int rb = m0 + wm * 32 + t * 16 + hi * 8;
            float fv[8];
#pragma unroll
            for (int r = 0; r < 8; ++r) fv[r] = fm[(rb + r) * NEXP + e];
#pragma unroll
            for (int u = 0; u < 4; ++u)
#pragma unroll
                for (int r = 0; r < 8; ++r)
                    tot[t][u][r] += fv[r] * acc[t][u][r];
        }
    }

    // epilogue: H = bf16( relu(hs)^2 )  (C layout: row = r + 8*hi, col = lane&15)
#pragma unroll
    for (int t = 0; t < 2; ++t) {
        int rb = m0 + wm * 32 + t * 16 + hi * 8;
#pragma unroll
        for (int u = 0; u < 4; ++u) {
            int col = i0 + wn * 64 + u * 16 + rsel;
#pragma unroll
            for (int r = 0; r < 8; ++r) {
                float h = fmaxf(tot[t][u][r], 0.f);
                H[(size_t)(rb + r) * IDIM + col] = (bf16)(h * h);
            }
        }
    }
}

// -------- Kernel 4: out = H @ out_w.T  (M=8192, N=1024, K=2048), f32 output --------
__global__ __launch_bounds__(256)
void moe_out_gemm(const bf16* __restrict__ H,
                  const bf16* __restrict__ owh,
                  float* __restrict__ out)
{
    const int tid  = threadIdx.x;
    const int lane = tid & 31;
    const int wv   = tid >> 5;
    const int wm   = wv >> 1;
    const int wn   = wv & 1;
    const int n0   = blockIdx.x * 128;             // over D
    const int m0   = blockIdx.y * 128;             // over tokens

    const int rsel  = lane & 15;
    const int hi    = lane >> 4;
    const int koffA = hi ? 8 : 0;
    const int koffB = hi ? 8 : 0;

    v8f acc[2][4];
#pragma unroll
    for (int t = 0; t < 2; ++t)
#pragma unroll
        for (int u = 0; u < 4; ++u)
            acc[t][u] = (v8f){0,0,0,0,0,0,0,0};

    for (int kb = 0; kb < IDIM; kb += 32) {
        v16bf a[2];
#pragma unroll
        for (int t = 0; t < 2; ++t) {
            const bf16* p = H + (size_t)(m0 + wm * 32 + t * 16 + rsel) * IDIM + kb + koffA;
            v8bf lo = *(const v8bf*)p;
            v8bf hh = *(const v8bf*)(p + 16);
            a[t] = __builtin_shufflevector(lo, hh, 0,1,2,3,4,5,6,7,8,9,10,11,12,13,14,15);
        }
        v16bf b[4];
#pragma unroll
        for (int u = 0; u < 4; ++u) {
            const bf16* p = owh + (size_t)(n0 + wn * 64 + u * 16 + rsel) * IDIM + kb + koffB;
            v8bf blo = *(const v8bf*)p;
            v8bf bhi = *(const v8bf*)(p + 16);
            b[u] = __builtin_shufflevector(blo, bhi, 0,1,2,3,4,5,6,7,8,9,10,11,12,13,14,15);
        }
#pragma unroll
        for (int t = 0; t < 2; ++t)
#pragma unroll
            for (int u = 0; u < 4; ++u)
                acc[t][u] = __builtin_amdgcn_wmma_f32_16x16x32_bf16(
                    false, a[t], false, b[u], (short)0, acc[t][u], false, false);
    }

#pragma unroll
    for (int t = 0; t < 2; ++t) {
        int rb = m0 + wm * 32 + t * 16 + hi * 8;
#pragma unroll
        for (int u = 0; u < 4; ++u) {
            int col = n0 + wn * 64 + u * 16 + rsel;
#pragma unroll
            for (int r = 0; r < 8; ++r)
                out[(size_t)(rb + r) * DDIM + col] = acc[t][u][r];
        }
    }
}

extern "C" void kernel_launch(void* const* d_in, const int* in_sizes, int n_in,
                              void* d_out, int out_size, void* d_ws, size_t ws_size,
                              hipStream_t stream)
{
    const float* x  = (const float*)d_in[0];   // (4,2048,1024)
    const float* gw = (const float*)d_in[1];   // (8,1024)
    const float* ew = (const float*)d_in[2];   // (8,2048,1024)
    const float* ow = (const float*)d_in[3];   // (1024,2048)
    float* out = (float*)d_out;

    // workspace layout (bytes)
    char* ws = (char*)d_ws;
    bf16*  xh  = (bf16*)(ws);                               // 16,777,216
    bf16*  ewh = (bf16*)(ws + 16777216);                    // 33,554,432
    bf16*  owh = (bf16*)(ws + 50331648);                    //  4,194,304
    float* fm  = (float*)(ws + 54525952);                   //    262,144
    bf16*  Hb  = (bf16*)(ws + 54788096);                    // 33,554,432  (total ~88.3 MB)

    // 1) gate + top-k weights + x -> bf16
    moe_gate_topk_cvt<<<N_TOK / 8, 256, 0, stream>>>(x, gw, xh, fm);

    // 2) weight converts
    cvt_f32_bf16<<<(NEXP * IDIM * DDIM / 4 + 255) / 256, 256, 0, stream>>>(ew, ewh, NEXP * IDIM * DDIM / 4);
    cvt_f32_bf16<<<(DDIM * IDIM / 4 + 255) / 256, 256, 0, stream>>>(ow, owh, DDIM * IDIM / 4);

    // 3) dense 8-expert WMMA GEMM with masked accumulate -> H = bf16(relu(hs)^2)
    {
        dim3 grid(IDIM / NT, N_TOK / MT);   // 16 x 64
        moe_expert_gemm<<<grid, 256, 0, stream>>>(xh, ewh, fm, Hb);
    }

    // 4) output WMMA GEMM -> f32 out
    {
        dim3 grid(DDIM / 128, N_TOK / 128); // 8 x 64
        moe_out_gemm<<<grid, 256, 0, stream>>>(Hb, owh, out);
    }
}
